// DQFunc_10144712753243
// MI455X (gfx1250) — matrix operand order repaired and checked by reading, not compiled, not run on hardware
//
#include <hip/hip_runtime.h>

typedef __attribute__((ext_vector_type(16))) __bf16 v16bf;
typedef __attribute__((ext_vector_type(8)))  float  v8f;

#define DEV __device__ __forceinline__

// Native converts: let the compiler emit v_cvt_*_bf16 instead of manual bit ops.
DEV __bf16 f2bf(float f)  { return (__bf16)f; }
DEV float  bf2f(__bf16 h) { return (float)h;  }

// A-fragment (16x32 bf16, MxK): lane<16: row M=lane, elems 0..7 = K kbase+half*8..+7,
// elems 8..15 = K kbase+16+half*8..+7 (per CDNA5 ISA 7.12.2).
DEV v16bf load_A_row(const __bf16* rowp, int kbase, int half) {
    union { __bf16 u[16]; v16bf v; } t;
    const __bf16* p = rowp + kbase + half * 8;
#pragma unroll
    for (int j = 0; j < 8; ++j) { t.u[j] = p[j]; t.u[8 + j] = p[16 + j]; }
    return t.v;
}

// B-fragment from packed weights: each (kt,nt) block is 32 lanes x 16 bf16 contiguous,
// so a lane reads 32 contiguous bytes -> two global_load_b128.
template<int NT>
DEV v16bf load_B_frag(const __bf16* W, int kt, int nt, int lane) {
    return *(const v16bf*)(W + (((size_t)(kt * NT + nt) * 32 + lane) * 16));
}

// D = A*B (+bias, act) ; A in LDS row-major bf16, output bf16 to LDS.
// ACT: 0=none 1=relu 2=tanh
template<int MT, int KT, int NT, int ACT>
DEV void gemm_lds(const __bf16* A, int lda,
                  const __bf16* W, const float* bias,
                  __bf16* O, int ldo, int wid, int lane) {
    const int half = lane >> 4, mr = lane & 15;
    for (int t = wid; t < MT * NT; t += 8) {
        const int mt = t / NT, nt = t % NT;
        const int m  = mt * 16 + mr;
        v8f acc = {};
#pragma unroll
        for (int kt = 0; kt < KT; ++kt) {
            v16bf a = load_A_row(A + m * lda, kt * 32, half);
            v16bf b = load_B_frag<NT>(W, kt, nt, lane);
            acc = __builtin_amdgcn_wmma_f32_16x16x32_bf16(false, a, false, b,
                                                          (short)0, acc, false, false);
        }
        const float bn = bias[nt * 16 + mr];
#pragma unroll
        for (int r = 0; r < 8; ++r) {
            float v = acc[r] + bn;
            if (ACT == 1) v = fmaxf(v, 0.f);
            if (ACT == 2) v = tanhf(v);
            O[(mt * 16 + half * 8 + r) * ldo + nt * 16 + mr] = f2bf(v);
        }
    }
}

// Attention layer 1: A = concat(h1[row], mean[row/20]) along K (512), no materialized concat.
template<int MT, int KT, int NT, int ACT>
DEV void gemm_att(const __bf16* H1, const __bf16* Mean,
                  const __bf16* W, const float* bias,
                  __bf16* O, int ldo, int wid, int lane) {
    const int half = lane >> 4, mr = lane & 15;
    for (int t = wid; t < MT * NT; t += 8) {
        const int mt = t / NT, nt = t % NT;
        const int m  = mt * 16 + mr;
        const __bf16* r0 = H1 + m * 256;
        const __bf16* r1 = Mean + (m / 20) * 256;
        v8f acc = {};
#pragma unroll
        for (int kt = 0; kt < KT; ++kt) {
            v16bf a = (kt < 8) ? load_A_row(r0, kt * 32, half)
                               : load_A_row(r1, kt * 32 - 256, half);
            v16bf b = load_B_frag<NT>(W, kt, nt, lane);
            acc = __builtin_amdgcn_wmma_f32_16x16x32_bf16(false, a, false, b,
                                                          (short)0, acc, false, false);
        }
        const float bn = bias[nt * 16 + mr];
#pragma unroll
        for (int r = 0; r < 8; ++r) {
            float v = acc[r] + bn;
            if (ACT == 1) v = fmaxf(v, 0.f);
            O[(mt * 16 + half * 8 + r) * ldo + nt * 16 + mr] = f2bf(v);
        }
    }
}

// Narrow-output GEMM (N<=16, single N-tile), f32 output, only first nvalid cols stored.
template<int MT, int KT>
DEV void gemm_f32(const __bf16* A, int lda,
                  const __bf16* W, const float* bias, int nvalid,
                  float* O, int ldo, int wid, int lane) {
    const int half = lane >> 4, mr = lane & 15;
    for (int mt = wid; mt < MT; mt += 8) {
        const int m = mt * 16 + mr;
        v8f acc = {};
#pragma unroll
        for (int kt = 0; kt < KT; ++kt) {
            v16bf a = load_A_row(A + m * lda, kt * 32, half);
            v16bf b = load_B_frag<1>(W, kt, 0, lane);
            acc = __builtin_amdgcn_wmma_f32_16x16x32_bf16(false, a, false, b,
                                                          (short)0, acc, false, false);
        }
        if (mr < nvalid) {
            const float bn = bias[mr];
#pragma unroll
            for (int r = 0; r < 8; ++r)
                O[(mt * 16 + half * 8 + r) * ldo + mr] = acc[r] + bn;
        }
    }
}

// ------- packed-weight offsets in d_ws (bf16 elements) -------
// layout per matrix: Ktiles x Ntiles blocks of 512 bf16 (lane-major fragments)
constexpr int O_M1W1 = 0;        // 13x128  Kt1  Nt8   4096
constexpr int O_M1W2 = 4096;     // 128x128 Kt4  Nt8   16384
constexpr int O_M1W3 = 20480;    // 128x256 Kt4  Nt16  32768
constexpr int O_M2W1 = 53248;    // 256x128 Kt8  Nt8   32768
constexpr int O_M2W2 = 86016;    // 128x128 Kt4  Nt8   16384
constexpr int O_M2W3 = 102400;   // 128x8   Kt4  Nt1   2048
constexpr int O_AW1  = 104448;   // 512x128 Kt16 Nt8   65536
constexpr int O_AW2  = 169984;   // 128x128 Kt4  Nt8   16384
constexpr int O_AW3  = 186368;   // 128x1   Kt4  Nt1   2048
constexpr int O_L1   = 188416;   // 16x256  Kt1  Nt16  8192
constexpr int O_L2   = 196608;   // 256x256 Kt8  Nt16  65536
// total 262144 bf16 = 512 KB of workspace

__global__ void pack_w(const float* __restrict__ W, __bf16* __restrict__ dst,
                       int K, int N, int NT, int total) {
    int idx = blockIdx.x * blockDim.x + threadIdx.x;
    if (idx >= total) return;
    int blk = idx >> 9, r = idx & 511;
    int lane = r >> 4, j = r & 15;
    int kt = blk / NT, nt = blk % NT;
    int half = lane >> 4;
    int n = nt * 16 + (lane & 15);
    int k = kt * 32 + (j < 8 ? half * 8 + j : 16 + half * 8 + (j - 8));   // B-layout: elems 0..7 K=half*8+j, elems 8..15 K=16+half*8+j-8
    float v = (k < K && n < N) ? W[(size_t)k * N + n] : 0.f;
    dst[idx] = f2bf(v);
}

struct FusedArgs {
    const float* state; const float* action;
    const __bf16* wp;
    const float *bm11, *bm12, *bm13;
    const float *bm21, *bm22, *bm23;
    const float *ba1, *ba2, *ba3;
    const float *bl1, *bl2;
    const float *q1w, *q1b, *q2w, *q2b;
    float* out; int B;
};

// One workgroup = 4 batch items = 80 rows = 5 M-tiles. 8 wave32.
__global__ __launch_bounds__(256, 1) void fused_fwd(FusedArgs p) {
    __shared__ __bf16 sA[80 * 32];     //  5 KB  state, K padded to 32
    __shared__ __bf16 sH[80 * 128];    // 20 KB  hidden ping
    __shared__ __bf16 sG[80 * 128];    // 20 KB  hidden pong
    __shared__ __bf16 sH1[80 * 256];   // 40 KB  mlp1 output
    __shared__ __bf16 sMean[4 * 256];  //  2 KB  per-item mean of h1
    __shared__ float  sFeat[80 * 8];   //  2.5 KB
    __shared__ float  sScore[80];
    __shared__ __bf16 sX[16 * 32];     //  1 KB  head input (4 rows live, padded)
    __shared__ __bf16 sXH[16 * 256];   //  8 KB
    __shared__ __bf16 sXH2[16 * 256];  //  8 KB

    const int tid   = threadIdx.x;
    const int wid   = tid >> 5, lane = tid & 31;
    const int bbase = blockIdx.x * 4;

    for (int i = tid; i < 80 * 32; i += 256) sA[i] = (__bf16)0.f;
    for (int i = tid; i < 16 * 32; i += 256) sX[i] = (__bf16)0.f;
    __syncthreads();
    for (int i = tid; i < 80 * 13; i += 256) {
        int row = i / 13, col = i % 13;
        sA[row * 32 + col] = f2bf(p.state[(size_t)(bbase * 20 + row) * 13 + col]);
    }
    __syncthreads();

    const __bf16* wp = p.wp;

    // ---- mlp1 : 13 -> 128 -> 128 -> 256 ----
    gemm_lds<5, 1, 8, 1>(sA, 32,  wp + O_M1W1, p.bm11, sH, 128, wid, lane); __syncthreads();
    gemm_lds<5, 4, 8, 1>(sH, 128, wp + O_M1W2, p.bm12, sG, 128, wid, lane); __syncthreads();
    gemm_lds<5, 4, 16, 0>(sG, 128, wp + O_M1W3, p.bm13, sH1, 256, wid, lane); __syncthreads();

    // ---- per-item mean of h1 over N=20 (256 threads = 256 cols) ----
    {
        const int c = tid;
        for (int it = 0; it < 4; ++it) {
            float s = 0.f;
            for (int n = 0; n < 20; ++n) s += bf2f(sH1[(it * 20 + n) * 256 + c]);
            sMean[it * 256 + c] = f2bf(s * 0.05f);
        }
    }
    __syncthreads();

    // ---- mlp2 : 256 -> 128 -> 128 -> 8 ----
    gemm_lds<5, 8, 8, 1>(sH1, 256, wp + O_M2W1, p.bm21, sH, 128, wid, lane); __syncthreads();
    gemm_lds<5, 4, 8, 1>(sH, 128,  wp + O_M2W2, p.bm22, sG, 128, wid, lane); __syncthreads();
    gemm_f32<5, 4>(sG, 128, wp + O_M2W3, p.bm23, 8, sFeat, 8, wid, lane);    __syncthreads();

    // ---- attention : concat(h1, mean) 512 -> 128 -> 128 -> 1 ----
    gemm_att<5, 16, 8, 1>(sH1, sMean, wp + O_AW1, p.ba1, sH, 128, wid, lane); __syncthreads();
    gemm_lds<5, 4, 8, 1>(sH, 128, wp + O_AW2, p.ba2, sG, 128, wid, lane);     __syncthreads();
    gemm_f32<5, 4>(sG, 128, wp + O_AW3, p.ba3, 1, sScore, 1, wid, lane);      __syncthreads();

    // ---- softmax over N=20 + weighted feature + assemble head input (wave per item) ----
    if (wid < 4) {
        const int it = wid;
        float s = (lane < 20) ? sScore[it * 20 + lane] : -3.0e38f;
        float mx = s;
        for (int o = 16; o; o >>= 1) mx = fmaxf(mx, __shfl_xor(mx, o, 32));
        float e = (lane < 20) ? __expf(s - mx) : 0.f;
        float sum = e;
        for (int o = 16; o; o >>= 1) sum += __shfl_xor(sum, o, 32);
        float w = e / sum;
        float wf[8];
#pragma unroll
        for (int c = 0; c < 8; ++c) {
            float pv = (lane < 20) ? w * sFeat[(it * 20 + lane) * 8 + c] : 0.f;
            for (int o = 16; o; o >>= 1) pv += __shfl_xor(pv, o, 32);
            wf[c] = pv;
        }
        if (lane == 0) {
            const int b = bbase + it;
#pragma unroll
            for (int c = 0; c < 6; ++c)
                sX[it * 32 + c] = f2bf(p.state[(size_t)b * 20 * 13 + c]);  // self_state
#pragma unroll
            for (int c = 0; c < 8; ++c) sX[it * 32 + 6 + c] = f2bf(wf[c]);
            sX[it * 32 + 14] = f2bf(p.action[b * 2 + 0]);
            sX[it * 32 + 15] = f2bf(p.action[b * 2 + 1]);
        }
    }
    __syncthreads();

    // ---- head : 16 -> 256 (tanh) -> 256 (tanh) -> {q1,q2} ----
    gemm_lds<1, 1, 16, 2>(sX, 32,   wp + O_L1, p.bl1, sXH, 256, wid, lane);  __syncthreads();
    gemm_lds<1, 8, 16, 2>(sXH, 256, wp + O_L2, p.bl2, sXH2, 256, wid, lane); __syncthreads();

    if (tid < 8) {
        const int it = tid >> 1, head = tid & 1;
        const float* qw = head ? p.q2w : p.q1w;
        const float  qb = head ? p.q2b[0] : p.q1b[0];
        float s = qb;
        for (int k = 0; k < 256; ++k) s += bf2f(sXH2[it * 256 + k]) * qw[k];
        p.out[(size_t)head * p.B + bbase + it] = s;
    }
}

extern "C" void kernel_launch(void* const* d_in, const int* in_sizes, int n_in,
                              void* d_out, int out_size, void* d_ws, size_t ws_size,
                              hipStream_t stream) {
    (void)n_in; (void)out_size; (void)ws_size;
    const float* state  = (const float*)d_in[0];
    const float* action = (const float*)d_in[1];
    const float* m1w1 = (const float*)d_in[2];  const float* m1b1 = (const float*)d_in[3];
    const float* m1w2 = (const float*)d_in[4];  const float* m1b2 = (const float*)d_in[5];
    const float* m1w3 = (const float*)d_in[6];  const float* m1b3 = (const float*)d_in[7];
    const float* m2w1 = (const float*)d_in[8];  const float* m2b1 = (const float*)d_in[9];
    const float* m2w2 = (const float*)d_in[10]; const float* m2b2 = (const float*)d_in[11];
    const float* m2w3 = (const float*)d_in[12]; const float* m2b3 = (const float*)d_in[13];
    const float* aw1  = (const float*)d_in[14]; const float* ab1  = (const float*)d_in[15];
    const float* aw2  = (const float*)d_in[16]; const float* ab2  = (const float*)d_in[17];
    const float* aw3  = (const float*)d_in[18]; const float* ab3  = (const float*)d_in[19];
    const float* l1w  = (const float*)d_in[20]; const float* l1b  = (const float*)d_in[21];
    const float* l2w  = (const float*)d_in[22]; const float* l2b  = (const float*)d_in[23];
    const float* q1w  = (const float*)d_in[24]; const float* q1b  = (const float*)d_in[25];
    const float* q2w  = (const float*)d_in[26]; const float* q2b  = (const float*)d_in[27];

    __bf16* wp = (__bf16*)d_ws;   // needs 512 KB of workspace
    auto packLaunch = [&](const float* W, int K, int N, int NT, int off, int sz) {
        pack_w<<<(sz + 255) / 256, 256, 0, stream>>>(W, wp + off, K, N, NT, sz);
    };
    packLaunch(m1w1, 13, 128, 8,   O_M1W1, 4096);
    packLaunch(m1w2, 128, 128, 8,  O_M1W2, 16384);
    packLaunch(m1w3, 128, 256, 16, O_M1W3, 32768);
    packLaunch(m2w1, 256, 128, 8,  O_M2W1, 32768);
    packLaunch(m2w2, 128, 128, 8,  O_M2W2, 16384);
    packLaunch(m2w3, 128, 8, 1,    O_M2W3, 2048);
    packLaunch(aw1,  512, 128, 8,  O_AW1,  65536);
    packLaunch(aw2,  128, 128, 8,  O_AW2,  16384);
    packLaunch(aw3,  128, 1, 1,    O_AW3,  2048);
    packLaunch(l1w,  16, 256, 16,  O_L1,   8192);
    packLaunch(l2w,  256, 256, 16, O_L2,   65536);

    const int B = in_sizes[1] / 2;   // action is (B, 2)
    FusedArgs a;
    a.state = state; a.action = action; a.wp = wp;
    a.bm11 = m1b1; a.bm12 = m1b2; a.bm13 = m1b3;
    a.bm21 = m2b1; a.bm22 = m2b2; a.bm23 = m2b3;
    a.ba1 = ab1; a.ba2 = ab2; a.ba3 = ab3;
    a.bl1 = l1b; a.bl2 = l2b;
    a.q1w = q1w; a.q1b = q1b; a.q2w = q2w; a.q2b = q2b;
    a.out = (float*)d_out; a.B = B;

    fused_fwd<<<B / 4, 256, 0, stream>>>(a);
}
